// GPTMoEParallelTransformerLayer_55336358642468
// MI455X (gfx1250) — compile-verified, matrix operand order checked by reading, not run on hardware
//
#include <hip/hip_runtime.h>
#include <hip/hip_bf16.h>

typedef __attribute__((ext_vector_type(16))) _Float16 v16h;
typedef __attribute__((ext_vector_type(8)))  _Float16 v8h;
typedef __attribute__((ext_vector_type(8)))  float    v8f;

#define S_LEN  1024
#define BATCH  4
#define HDIM   768
#define NHEAD  12
#define HEADD  64
#define FFDIM  3072
#define NEXP   4
#define NTOK   4096   // S_LEN * BATCH, token index t = s*BATCH + b

// GEMM epilogue flags
#define F_BIAS  1
#define F_GELU  2
#define F_RES   4
#define F_OUT32 8
#define F_OUT16 16
#define F_GACC  32
#define F_OUT2  64

// ---------------------------------------------------------------------------
// CDNA5 async global->LDS copy (ASYNCcnt path). Per-lane 16B transfer, data
// never touches VGPRs.  dsaddr = low 32 bits of the generic LDS pointer.
// ---------------------------------------------------------------------------
__device__ __forceinline__ void async_load_b128(void* lds_ptr, const void* gptr) {
  const unsigned int l = (unsigned int)(uintptr_t)lds_ptr;
  asm volatile("global_load_async_to_lds_b128 %0, %1, off"
               :: "v"(l), "v"(gptr) : "memory");
}
__device__ __forceinline__ void wait_async0() {
  asm volatile("s_wait_asynccnt 0x0" ::: "memory");
}

// ---------------------------------------------------------------------------
// Fragment loader: 16x32 f16 A-tile (or W[N,K] row-major B-tile via symmetry).
// Lane layout per CDNA5 ISA 7.12.2: lanes 0-15 row M=lane hold K=0..7,16..23;
// lanes 16-31 row M=lane-16 hold K=8..15,24..31.  Two 16B loads per fragment.
// ---------------------------------------------------------------------------
__device__ __forceinline__ v16h load_frag(const _Float16* p, int ld, int lane) {
  const int m  = lane & 15;
  const int ko = (lane >> 4) << 3;                 // 0 or 8
  const _Float16* q = p + m * ld + ko;
  v8h lo = *(const v8h*)(q);                       // K = ko .. ko+7
  v8h hi = *(const v8h*)(q + 16);                  // K = 16+ko .. 16+ko+7
  return __builtin_shufflevector(lo, hi, 0, 1, 2, 3, 4, 5, 6, 7,
                                 8, 9, 10, 11, 12, 13, 14, 15);
}

// ---------------------------------------------------------------------------
// Generic WMMA GEMM: C[M,N] = A[M,K](f16) * W[N,K](f16)^T, fused epilogue.
// 256 threads = 8 waves, wave (wm in 0..3, wn in 0..1) owns a 32x64 tile
// (8 accumulators; each B fragment feeds two WMMAs) -> block tile 128x128.
// A (128x32) and B (128x32) K-slices staged in LDS by double-buffered async
// copies: copies for slice s+1 overlap the WMMAs of slice s.  All operand
// fragments are hoisted ahead of the WMMA burst so LDS latency is amortized
// over one load clause instead of per-pair waits.
// M%128==0, N%128==0, K%32==0 at all call sites.
// ---------------------------------------------------------------------------
__global__ __launch_bounds__(256) void wmma_gemm_kernel(
    const _Float16* __restrict__ A, const _Float16* __restrict__ W,
    const float* __restrict__ bias, const float* __restrict__ res,
    float* __restrict__ out32, float* __restrict__ out2,
    _Float16* __restrict__ out16,
    const int* __restrict__ eidx, const float* __restrict__ prob, int expert,
    int M, int N, int K, int flags)
{
  __shared__ __align__(16) _Float16 sA[2][128 * 32];   // 8 KB per buffer
  __shared__ __align__(16) _Float16 sB[2][128 * 32];   // 8 KB per buffer

  const int tid  = threadIdx.x;
  const int lane = tid & 31;
  const int wave = tid >> 5;
  const int wm   = wave & 3;            // 32-row band
  const int wn   = wave >> 2;           // 64-col half
  const int mBlk = blockIdx.y * 128;
  const int nBlk = blockIdx.x * 128;
  (void)M;

  // Per-thread async copy slots: two 16B chunks for A, two for B.
  const int idx0 = tid * 8;             // halves index into a 128x32 tile
  const int idx1 = tid * 8 + 2048;
  const int r0 = idx0 >> 5, k0 = idx0 & 31;
  const int r1 = idx1 >> 5, k1 = idx1 & 31;

  const _Float16* Ablk = A + (size_t)mBlk * K;

  auto issue = [&](int buf, int kb) {
    async_load_b128(&sA[buf][idx0], Ablk + (size_t)r0 * K + kb + k0);
    async_load_b128(&sA[buf][idx1], Ablk + (size_t)r1 * K + kb + k1);
    async_load_b128(&sB[buf][idx0], W + (size_t)(nBlk + r0) * K + kb + k0);
    async_load_b128(&sB[buf][idx1], W + (size_t)(nBlk + r1) * K + kb + k1);
  };

  v8f acc0[4] = {};
  v8f acc1[4] = {};
  const int nsteps = K >> 5;
  issue(0, 0);
  for (int s = 0; s < nsteps; ++s) {
    wait_async0();                      // my copies for slice s are done
    __syncthreads();                    // everyone's copies done; prev reads done
    if (s + 1 < nsteps) issue((s + 1) & 1, (s + 1) << 5);
    const int buf = s & 1;
    const _Float16* aB = &sA[buf][(wm * 32) * 32];
    const _Float16* bB = &sB[buf][(wn * 64) * 32];
    // hoist all operand fragments: one load clause, amortized dscnt waits
    v16h a0 = load_frag(aB,           32, lane);
    v16h a1 = load_frag(aB + 16 * 32, 32, lane);
    v16h b0 = load_frag(bB +  0 * 16 * 32, 32, lane);
    v16h b1 = load_frag(bB +  1 * 16 * 32, 32, lane);
    v16h b2 = load_frag(bB +  2 * 16 * 32, 32, lane);
    v16h b3 = load_frag(bB +  3 * 16 * 32, 32, lane);
    acc0[0] = __builtin_amdgcn_wmma_f32_16x16x32_f16(false, a0, false, b0, (short)0, acc0[0], false, false);
    acc1[0] = __builtin_amdgcn_wmma_f32_16x16x32_f16(false, a1, false, b0, (short)0, acc1[0], false, false);
    acc0[1] = __builtin_amdgcn_wmma_f32_16x16x32_f16(false, a0, false, b1, (short)0, acc0[1], false, false);
    acc1[1] = __builtin_amdgcn_wmma_f32_16x16x32_f16(false, a1, false, b1, (short)0, acc1[1], false, false);
    acc0[2] = __builtin_amdgcn_wmma_f32_16x16x32_f16(false, a0, false, b2, (short)0, acc0[2], false, false);
    acc1[2] = __builtin_amdgcn_wmma_f32_16x16x32_f16(false, a1, false, b2, (short)0, acc1[2], false, false);
    acc0[3] = __builtin_amdgcn_wmma_f32_16x16x32_f16(false, a0, false, b3, (short)0, acc0[3], false, false);
    acc1[3] = __builtin_amdgcn_wmma_f32_16x16x32_f16(false, a1, false, b3, (short)0, acc1[3], false, false);
  }

  const int half = lane >> 4;
  const int col0 = lane & 15;
  const int nBase = nBlk + wn * 64;
#pragma unroll
  for (int mt = 0; mt < 2; ++mt) {
    const int mBase = mBlk + wm * 32 + mt * 16;
    v8f* acc = mt ? acc1 : acc0;
#pragma unroll
    for (int j = 0; j < 4; ++j) {
      const int col = nBase + 16 * j + col0;
#pragma unroll
      for (int r = 0; r < 8; ++r) {
        const int row = mBase + r + 8 * half;   // C layout: vgpr r -> M=r / r+8
        float v = acc[j][r];
        if (flags & F_BIAS) v += bias[col];
        if (flags & F_GELU) v = 0.5f * v * (1.0f + erff(v * 0.70710678118f));
        if (flags & F_RES)  v += res[(size_t)row * N + col];
        if (flags & F_GACC) {
          const float g = (eidx[row] == expert) ? prob[row] : 0.0f;
          out32[(size_t)row * N + col] += g * v;
        } else {
          if (flags & F_OUT32) out32[(size_t)row * N + col] = v;
          if (flags & F_OUT2)  out2[(size_t)row * N + col] = v;
        }
        if (flags & F_OUT16) out16[(size_t)row * N + col] = (_Float16)v;
      }
    }
  }
}

// ---------------------------------------------------------------------------
// Flash attention, causal. Grid: (S/64 q-tiles, BATCH*NHEAD heads), 128 thr.
// Q pre-scaled by 1/8. Q/K: [head][s][64], V transposed: [head][64][s].
// K/V tiles staged in LDS via async copies; P converted C-layout -> A-layout
// through LDS.
// ---------------------------------------------------------------------------
__global__ __launch_bounds__(128) void attn_kernel(
    const _Float16* __restrict__ Q, const _Float16* __restrict__ Kd,
    const _Float16* __restrict__ Vt, _Float16* __restrict__ ctx)
{
  __shared__ __align__(16) _Float16 sK[64 * 64];
  __shared__ __align__(16) _Float16 sV[64 * 64];
  __shared__ __align__(16) _Float16 sP[4 * 16 * 64];

  const int tid  = threadIdx.x;
  const int lane = tid & 31;
  const int wave = tid >> 5;            // 0..3 -> 16-row band
  const int half = lane >> 4;
  const int col0 = lane & 15;
  const int qt   = blockIdx.x;          // query tile (64 rows)
  const int head = blockIdx.y;          // b*NHEAD + n
  const int b    = head / NHEAD;
  const int nh   = head % NHEAD;

  const size_t headOff = (size_t)head * S_LEN * HEADD;

  const _Float16* qPtr = Q + headOff + (size_t)(qt * 64 + wave * 16) * HEADD;
  v16h qf0 = load_frag(qPtr + 0,  HEADD, lane);
  v16h qf1 = load_frag(qPtr + 32, HEADD, lane);

  v8f o[4] = {};
  float mrow[8], lrow[8];
#pragma unroll
  for (int r = 0; r < 8; ++r) { mrow[r] = -1e30f; lrow[r] = 0.0f; }

  // V-tile staging geometry (per thread): row d, 32-half segment.
  const int vd   = tid >> 1;
  const int voff = (tid & 1) * 32;

  for (int kvt = 0; kvt <= qt; ++kvt) {
    const int kvBase = kvt * 64;
    __syncthreads();                    // previous tile fully consumed
    {   // async-stage K tile [64 kv][64 hd] and V tile [64 hd][64 kv]
      const _Float16* ksrc = Kd + headOff + (size_t)kvBase * HEADD;
#pragma unroll
      for (int c = 0; c < 4; ++c) {
        const int i = tid * 8 + c * 1024;
        async_load_b128(sK + i, ksrc + i);
      }
      const _Float16* vsrc = Vt + headOff + (size_t)vd * S_LEN + kvBase + voff;
      _Float16* vdst = sV + vd * 64 + voff;
#pragma unroll
      for (int c = 0; c < 4; ++c)
        async_load_b128(vdst + c * 8, vsrc + c * 8);
    }
    wait_async0();
    __syncthreads();

    // S = Q K^T  (16x64 per wave, K=64 in two 32-steps); hoist both K frags
    v8f sacc[4] = {};
#pragma unroll
    for (int j = 0; j < 4; ++j) {
      v16h b0 = load_frag(sK + 16 * j * 64 + 0,  64, lane);
      v16h b1 = load_frag(sK + 16 * j * 64 + 32, 64, lane);
      sacc[j] = __builtin_amdgcn_wmma_f32_16x16x32_f16(
          false, qf0, false, b0, (short)0, sacc[j], false, false);
      sacc[j] = __builtin_amdgcn_wmma_f32_16x16x32_f16(
          false, qf1, false, b1, (short)0, sacc[j], false, false);
    }

    if (kvt == qt) {                    // causal mask on diagonal tile
#pragma unroll
      for (int j = 0; j < 4; ++j) {
        const int n = 16 * j + col0;
#pragma unroll
        for (int r = 0; r < 8; ++r) {
          const int m = wave * 16 + r + 8 * half;
          if (n > m) sacc[j][r] = -10000.0f;
        }
      }
    }

    // online softmax (row reductions across the 16-lane half-groups)
#pragma unroll
    for (int r = 0; r < 8; ++r) {
      float v = fmaxf(fmaxf(sacc[0][r], sacc[1][r]), fmaxf(sacc[2][r], sacc[3][r]));
      v = fmaxf(v, __shfl_xor(v, 1, 16));
      v = fmaxf(v, __shfl_xor(v, 2, 16));
      v = fmaxf(v, __shfl_xor(v, 4, 16));
      v = fmaxf(v, __shfl_xor(v, 8, 16));
      const float mnew = fmaxf(mrow[r], v);
      const float sc   = __expf(mrow[r] - mnew);
      mrow[r] = mnew;
      float rs = 0.0f;
#pragma unroll
      for (int j = 0; j < 4; ++j) {
        const float p = __expf(sacc[j][r] - mnew);
        sP[wave * 1024 + (r + 8 * half) * 64 + 16 * j + col0] = (_Float16)p;
        rs += p;
      }
      rs += __shfl_xor(rs, 1, 16);
      rs += __shfl_xor(rs, 2, 16);
      rs += __shfl_xor(rs, 4, 16);
      rs += __shfl_xor(rs, 8, 16);
      lrow[r] = lrow[r] * sc + rs;
#pragma unroll
      for (int j = 0; j < 4; ++j) o[j][r] *= sc;
    }

    // O += P * V  (K dim = 64 kv positions); hoist P + all V frags per kt
#pragma unroll
    for (int kt = 0; kt < 2; ++kt) {
      v16h pf  = load_frag(sP + wave * 1024 + kt * 32, 64, lane);
      v16h vf0 = load_frag(sV + 0 * 16 * 64 + kt * 32, 64, lane);
      v16h vf1 = load_frag(sV + 1 * 16 * 64 + kt * 32, 64, lane);
      v16h vf2 = load_frag(sV + 2 * 16 * 64 + kt * 32, 64, lane);
      v16h vf3 = load_frag(sV + 3 * 16 * 64 + kt * 32, 64, lane);
      o[0] = __builtin_amdgcn_wmma_f32_16x16x32_f16(false, pf, false, vf0, (short)0, o[0], false, false);
      o[1] = __builtin_amdgcn_wmma_f32_16x16x32_f16(false, pf, false, vf1, (short)0, o[1], false, false);
      o[2] = __builtin_amdgcn_wmma_f32_16x16x32_f16(false, pf, false, vf2, (short)0, o[2], false, false);
      o[3] = __builtin_amdgcn_wmma_f32_16x16x32_f16(false, pf, false, vf3, (short)0, o[3], false, false);
    }
  }

  // write context f16: ctx[t*HDIM + nh*64 + d], t = qpos*BATCH + b
#pragma unroll
  for (int j = 0; j < 4; ++j) {
    const int d = 16 * j + col0;
#pragma unroll
    for (int r = 0; r < 8; ++r) {
      const int qpos = qt * 64 + wave * 16 + r + 8 * half;
      const int t = qpos * BATCH + b;
      ctx[(size_t)t * HDIM + nh * 64 + d] = (_Float16)(o[j][r] / lrow[r]);
    }
  }
}

// ---------------------------------------------------------------------------
// LayerNorm (+optional top-1 gate): one block per token, 256 thr x 3 cols.
// ---------------------------------------------------------------------------
__global__ __launch_bounds__(256) void ln_gate_kernel(
    const float* __restrict__ x, const float* __restrict__ w,
    const float* __restrict__ bsh, _Float16* __restrict__ out16,
    const float* __restrict__ gw, int* __restrict__ eidx,
    float* __restrict__ prob)
{
  __shared__ float red[256];
  __shared__ float sstat[2];
  __shared__ float sg[NEXP];
  const int t = blockIdx.x;
  const int tid = threadIdx.x;
  const float* xr = x + (size_t)t * HDIM;
  float xv[3];
  float s = 0.0f, sq = 0.0f;
#pragma unroll
  for (int p = 0; p < 3; ++p) {
    xv[p] = xr[tid + 256 * p];
    s += xv[p]; sq += xv[p] * xv[p];
  }
  red[tid] = s; __syncthreads();
  for (int st = 128; st > 0; st >>= 1) { if (tid < st) red[tid] += red[tid + st]; __syncthreads(); }
  if (tid == 0) sstat[0] = red[0] * (1.0f / HDIM);
  __syncthreads();
  red[tid] = sq; __syncthreads();
  for (int st = 128; st > 0; st >>= 1) { if (tid < st) red[tid] += red[tid + st]; __syncthreads(); }
  if (tid == 0) {
    const float mean = sstat[0];
    const float var  = red[0] * (1.0f / HDIM) - mean * mean;
    sstat[1] = rsqrtf(var + 1e-5f);
  }
  __syncthreads();
  const float mean = sstat[0], rstd = sstat[1];
  float g[NEXP] = {0.0f, 0.0f, 0.0f, 0.0f};
#pragma unroll
  for (int p = 0; p < 3; ++p) {
    const int i = tid + 256 * p;
    const float y = (xv[p] - mean) * rstd * w[i] + bsh[i];
    out16[(size_t)t * HDIM + i] = (_Float16)y;
    if (gw) {
#pragma unroll
      for (int e = 0; e < NEXP; ++e) g[e] += y * gw[e * HDIM + i];
    }
  }
  if (gw) {
    for (int e = 0; e < NEXP; ++e) {
      __syncthreads();
      red[tid] = g[e]; __syncthreads();
      for (int st = 128; st > 0; st >>= 1) { if (tid < st) red[tid] += red[tid + st]; __syncthreads(); }
      if (tid == 0) sg[e] = red[0];
    }
    __syncthreads();
    if (tid == 0) {
      const float mx = fmaxf(fmaxf(sg[0], sg[1]), fmaxf(sg[2], sg[3]));
      float ex[NEXP], den = 0.0f;
      for (int e = 0; e < NEXP; ++e) { ex[e] = __expf(sg[e] - mx); den += ex[e]; }
      int best = 0; float bestv = -1.0f;
      for (int e = 0; e < NEXP; ++e) {
        const float pe = ex[e] / den;
        if (pe > bestv) { bestv = pe; best = e; }
      }
      eidx[t] = best;
      prob[t] = bestv;
    }
  }
}

// ---------------------------------------------------------------------------
// Elementwise helpers
// ---------------------------------------------------------------------------
__global__ void f32_to_f16_kernel(const float* __restrict__ in,
                                  _Float16* __restrict__ out, int n) {
  const int i = blockIdx.x * 256 + threadIdx.x;
  if (i < n) out[i] = (_Float16)in[i];
}

// in [nmat][rows][cols] f32 -> out [nmat][cols][rows] f16
__global__ void transpose_f32_to_f16_kernel(const float* __restrict__ in,
                                            _Float16* __restrict__ out,
                                            int rows, int cols, int nmat) {
  const int idx = blockIdx.x * 256 + threadIdx.x;
  const int per = rows * cols;
  if (idx >= per * nmat) return;
  const int mat = idx / per;
  const int rem = idx - mat * per;
  const int r = rem / cols;
  const int c = rem - r * cols;
  out[(size_t)mat * per + (size_t)c * rows + r] = (_Float16)in[idx];
}

// qkv f32 [t][3H] -> q16/k16 [head][s][64] (q scaled 1/8), vt16 [head][64][s]
__global__ void repack_qkv_kernel(const float* __restrict__ qkv,
                                  _Float16* __restrict__ q,
                                  _Float16* __restrict__ k,
                                  _Float16* __restrict__ vt) {
  const int idx = blockIdx.x * 256 + threadIdx.x;
  if (idx >= NTOK * HDIM) return;
  const int t = idx / HDIM;
  const int c = idx - t * HDIM;
  const int n = c >> 6;
  const int d = c & 63;
  const int s = t >> 2;          // t = s*BATCH + b
  const int b = t & 3;
  const int h = b * NHEAD + n;
  const float* base = qkv + (size_t)t * (3 * HDIM) + n * (3 * HEADD);
  q[((size_t)h * S_LEN + s) * HEADD + d]  = (_Float16)(base[d] * 0.125f);
  k[((size_t)h * S_LEN + s) * HEADD + d]  = (_Float16)(base[HEADD + d]);
  vt[((size_t)h * HEADD + d) * S_LEN + s] = (_Float16)(base[2 * HEADD + d]);
}

// ---------------------------------------------------------------------------
// Host orchestration
// ---------------------------------------------------------------------------
extern "C" void kernel_launch(void* const* d_in, const int* in_sizes, int n_in,
                              void* d_out, int out_size, void* d_ws, size_t ws_size,
                              hipStream_t stream) {
  (void)in_sizes; (void)n_in; (void)out_size; (void)ws_size;
  const float* hidden = (const float*)d_in[0];
  /* d_in[1] = attention_mask (bool) — causal mask is applied analytically */
  const float* ln1w = (const float*)d_in[2];
  const float* ln1b = (const float*)d_in[3];
  const float* qkvw = (const float*)d_in[4];
  const float* qkvb = (const float*)d_in[5];
  const float* dw   = (const float*)d_in[6];
  const float* db   = (const float*)d_in[7];
  const float* ln2w = (const float*)d_in[8];
  const float* ln2b = (const float*)d_in[9];
  const float* gw   = (const float*)d_in[10];
  const float* w1   = (const float*)d_in[11];
  const float* b1   = (const float*)d_in[12];
  const float* w2   = (const float*)d_in[13];
  const float* b2   = (const float*)d_in[14];

  char* ws = (char*)d_ws;
  size_t off = 0;
  auto alloc = [&](size_t bytes) -> void* {
    void* p = ws + off;
    off += (bytes + 255) & ~(size_t)255;
    return p;
  };

  _Float16* qkvw16 = (_Float16*)alloc((size_t)3 * HDIM * HDIM * 2);
  _Float16* dw16   = (_Float16*)alloc((size_t)HDIM * HDIM * 2);
  _Float16* w1t16  = (_Float16*)alloc((size_t)NEXP * HDIM * FFDIM * 2);
  _Float16* w2t16  = (_Float16*)alloc((size_t)NEXP * HDIM * FFDIM * 2);
  _Float16* ln1_16 = (_Float16*)alloc((size_t)NTOK * HDIM * 2);
  float*    qkv32  = (float*)alloc((size_t)NTOK * 3 * HDIM * 4);
  _Float16* q16    = (_Float16*)alloc((size_t)NTOK * HDIM * 2);
  _Float16* k16    = (_Float16*)alloc((size_t)NTOK * HDIM * 2);
  _Float16* vt16   = (_Float16*)alloc((size_t)NTOK * HDIM * 2);
  float*    x1     = (float*)alloc((size_t)NTOK * HDIM * 4);
  _Float16* ln2_16 = (_Float16*)alloc((size_t)NTOK * HDIM * 2);
  int*      eidx   = (int*)alloc((size_t)NTOK * 4);
  float*    prob   = (float*)alloc((size_t)NTOK * 4);
  _Float16* ctx16  = ln1_16;            // reuse: ln1 dead after QKV GEMM
  _Float16* h1_16  = (_Float16*)qkv32;  // reuse: qkv f32 dead after repack

  // --- weight precision conversion / transposes ---
  f32_to_f16_kernel<<<(3 * HDIM * HDIM + 255) / 256, 256, 0, stream>>>(
      qkvw, qkvw16, 3 * HDIM * HDIM);
  f32_to_f16_kernel<<<(HDIM * HDIM + 255) / 256, 256, 0, stream>>>(
      dw, dw16, HDIM * HDIM);
  transpose_f32_to_f16_kernel<<<(NEXP * HDIM * FFDIM + 255) / 256, 256, 0, stream>>>(
      w1, w1t16, HDIM, FFDIM, NEXP);     // [E,H,FF] -> [E,FF,H]
  transpose_f32_to_f16_kernel<<<(NEXP * HDIM * FFDIM + 255) / 256, 256, 0, stream>>>(
      w2, w2t16, FFDIM, HDIM, NEXP);     // [E,FF,H] -> [E,H,FF]

  // --- LN1 ---
  ln_gate_kernel<<<NTOK, 256, 0, stream>>>(hidden, ln1w, ln1b, ln1_16,
                                           nullptr, nullptr, nullptr);
  // --- QKV projection ---
  wmma_gemm_kernel<<<dim3(3 * HDIM / 128, NTOK / 128), 256, 0, stream>>>(
      ln1_16, qkvw16, qkvb, nullptr, qkv32, nullptr, nullptr,
      nullptr, nullptr, 0, NTOK, 3 * HDIM, HDIM, F_BIAS | F_OUT32);
  repack_qkv_kernel<<<(NTOK * HDIM + 255) / 256, 256, 0, stream>>>(
      qkv32, q16, k16, vt16);
  // --- attention ---
  attn_kernel<<<dim3(S_LEN / 64, BATCH * NHEAD), 128, 0, stream>>>(
      q16, k16, vt16, ctx16);
  // --- output dense + residual (x1 -> ws and d_out) ---
  wmma_gemm_kernel<<<dim3(HDIM / 128, NTOK / 128), 256, 0, stream>>>(
      ctx16, dw16, db, hidden, x1, (float*)d_out, nullptr,
      nullptr, nullptr, 0, NTOK, HDIM, HDIM,
      F_BIAS | F_RES | F_OUT32 | F_OUT2);
  // --- LN2 + top-1 gate ---
  ln_gate_kernel<<<NTOK, 256, 0, stream>>>(x1, ln2w, ln2b, ln2_16,
                                           gw, eidx, prob);
  // --- MoE experts: gelu(t @ w1 + b1) @ w2 + b2, gated accumulate ---
  for (int e = 0; e < NEXP; ++e) {
    wmma_gemm_kernel<<<dim3(FFDIM / 128, NTOK / 128), 256, 0, stream>>>(
        ln2_16, w1t16 + (size_t)e * HDIM * FFDIM, b1 + e * FFDIM, nullptr,
        nullptr, nullptr, h1_16, nullptr, nullptr, 0,
        NTOK, FFDIM, HDIM, F_BIAS | F_GELU | F_OUT16);
    wmma_gemm_kernel<<<dim3(HDIM / 128, NTOK / 128), 256, 0, stream>>>(
        h1_16, w2t16 + (size_t)e * HDIM * FFDIM, b2 + e * HDIM, nullptr,
        (float*)d_out, nullptr, nullptr, eidx, prob, e,
        NTOK, HDIM, FFDIM, F_BIAS | F_GACC);
  }
}
